// MessageActor_78280073936943
// MI455X (gfx1250) — matrix operand order repaired and checked by reading, not compiled, not run on hardware
//
#include <hip/hip_runtime.h>
#include <hip/hip_bf16.h>
#include <math.h>

// ---- problem constants (match reference) ----
#define BB   256
#define NAG  8192
#define NE   262144
#define NACT 21

typedef __attribute__((ext_vector_type(16))) _Float16 v16h;
typedef __attribute__((ext_vector_type(8)))  float    v8f;
typedef __attribute__((ext_vector_type(2)))  float    v2f;
typedef __attribute__((ext_vector_type(4)))  int      v4i;

#if defined(__has_builtin)
#  if __has_builtin(__builtin_amdgcn_wmma_f32_16x16x4_f32)
#    define HAVE_WMMA_F32X4 1
#  endif
#  if __has_builtin(__builtin_amdgcn_global_load_async_to_lds_b128) && \
      __has_builtin(__builtin_amdgcn_global_load_async_to_lds_b32)
#    define HAVE_ASYNC_LDS 1
#  endif
#  if __has_builtin(__builtin_amdgcn_s_wait_asynccnt)
#    define HAVE_WAIT_ASYNC 1
#  endif
#endif
#ifndef HAVE_WMMA_F32X4
#  define HAVE_WMMA_F32X4 0
#endif
#ifndef HAVE_ASYNC_LDS
#  define HAVE_ASYNC_LDS 0
#endif
#ifndef HAVE_WAIT_ASYNC
#  define HAVE_WAIT_ASYNC 0
#endif

#if HAVE_ASYNC_LDS
// Builtin params (from compiler diagnostics): typed pointers, AS(1) global
// source / AS(3) LDS destination. generic->AS(3) keeps the low 32 bits, which
// are the LDS offset per the ISA flat-aperture rule.
typedef __attribute__((address_space(1))) v4i* gp_v4i;
typedef __attribute__((address_space(3))) v4i* lp_v4i;
typedef __attribute__((address_space(1))) int* gp_i32;
typedef __attribute__((address_space(3))) int* lp_i32;
__device__ __forceinline__ gp_v4i as_g4(const void* p) { return (gp_v4i)(uintptr_t)p; }
__device__ __forceinline__ lp_v4i as_l4(const void* p) { return (lp_v4i)(unsigned)(uintptr_t)p; }
__device__ __forceinline__ gp_i32 as_g1(const void* p) { return (gp_i32)(uintptr_t)p; }
__device__ __forceinline__ lp_i32 as_l1(const void* p) { return (lp_i32)(unsigned)(uintptr_t)p; }
#endif

#if HAVE_WAIT_ASYNC
#  define WAIT_ASYNC(n) __builtin_amdgcn_s_wait_asynccnt(n)
#elif HAVE_ASYNC_LDS
#  define WAIT_ASYNC(n)                                                \
    do {                                                               \
      if ((n) == 0) asm volatile("s_wait_asynccnt 0x0" ::: "memory");  \
      else          asm volatile("s_wait_asynccnt 0x5" ::: "memory");  \
    } while (0)
#else
#  define WAIT_ASYNC(n) do {} while (0)
#endif

// ---------------------------------------------------------------------------
// conv1: x[256,5,13,13] * w[32,5,3,3] stride2 VALID -> relu -> h1[256,32,6,6]
// ---------------------------------------------------------------------------
__global__ void conv1_kernel(const float* __restrict__ x, const float* __restrict__ w,
                             const float* __restrict__ bias, float* __restrict__ out) {
  int t = blockIdx.x * blockDim.x + threadIdx.x;
  if (t >= BB * 32 * 36) return;
  int j = t % 6, i = (t / 6) % 6, o = (t / 36) % 32, b = t / (36 * 32);
  float acc = bias[o];
  const float* xb = x + b * 5 * 169;
  const float* wo = w + o * 45;
#pragma unroll
  for (int c = 0; c < 5; ++c)
#pragma unroll
    for (int u = 0; u < 3; ++u)
#pragma unroll
      for (int v = 0; v < 3; ++v)
        acc += xb[c * 169 + (2 * i + u) * 13 + (2 * j + v)] * wo[c * 9 + u * 3 + v];
  out[t] = fmaxf(acc, 0.f);
}

// ---------------------------------------------------------------------------
// conv2: h1[256,32,6,6] * w[64,32,3,3] stride1 -> relu -> h2[256,64,4,4]=[256,1024]
// ---------------------------------------------------------------------------
__global__ void conv2_kernel(const float* __restrict__ h1, const float* __restrict__ w,
                             const float* __restrict__ bias, float* __restrict__ out) {
  int t = blockIdx.x * blockDim.x + threadIdx.x;
  if (t >= BB * 64 * 16) return;
  int j = t % 4, i = (t / 4) % 4, o = (t / 16) % 64, b = t / 1024;
  float acc = bias[o];
  const float* hb = h1 + b * 32 * 36;
  const float* wo = w + o * 32 * 9;
  for (int c = 0; c < 32; ++c)
#pragma unroll
    for (int u = 0; u < 3; ++u)
#pragma unroll
      for (int v = 0; v < 3; ++v)
        acc += hb[c * 36 + (i + u) * 6 + (j + v)] * wo[c * 9 + u * 3 + v];
  out[t] = fmaxf(acc, 0.f);
}

// ---------------------------------------------------------------------------
// FC: hidden = relu(h2[256,1024] @ fc_w.T[1024,512] + fc_b), f16 WMMA 16x16x32.
//   A (16x32): lane<16 e->K in {0..7,16..23}; lane>=16 e->K in {8..15,24..31}
//   B (32x16): same K order as A; N = lane%16
// ---------------------------------------------------------------------------
__global__ void fc_wmma_kernel(const float* __restrict__ h2, const float* __restrict__ fw,
                               const float* __restrict__ fb, float* __restrict__ hidden) {
  int wave = (blockIdx.x * blockDim.x + threadIdx.x) >> 5;  // 0..511 tiles
  int lane = threadIdx.x & 31;
  int tm = wave & 15, tn = wave >> 4;  // M tiles: 16, N tiles: 32
  int half = lane >> 4, lid = lane & 15;
  int row = tm * 16 + lid;
  int col = tn * 16 + lid;
  v8f acc = {};
  for (int k0 = 0; k0 < 1024; k0 += 32) {
    v16h a, b;
#pragma unroll
    for (int e = 0; e < 16; ++e) {
      int K = (e < 8 ? e : e + 8) + half * 8;
      a[e] = (_Float16)h2[row * 1024 + k0 + K];
    }
#pragma unroll
    for (int e = 0; e < 16; ++e) {
      int K = (e < 8 ? e : e + 8) + half * 8;
      b[e] = (_Float16)fw[col * 1024 + k0 + K];  // B[k,n] = fc_w[n,k]
    }
    acc = __builtin_amdgcn_wmma_f32_16x16x32_f16(false, a, false, b, (short)0, acc,
                                                 false, false);
  }
  float bias = fb[col];
#pragma unroll
  for (int r = 0; r < 8; ++r) {
    int orow = tm * 16 + r + half * 8;  // C/D layout: VGPR r -> M = r + 8*half
    hidden[orow * 512 + col] = fmaxf(acc[r] + bias, 0.f);
  }
}

// ---------------------------------------------------------------------------
// Policy head: logits, log_prob(action), entropy; one wave per batch row.
// ---------------------------------------------------------------------------
__global__ void head_kernel(const float* __restrict__ hidden, const float* __restrict__ mw,
                            const float* __restrict__ mb, const int* __restrict__ action,
                            float* __restrict__ out) {
  int b = blockIdx.x;
  int j = threadIdx.x;  // blockDim = 32, lanes 0..20 are actions
  float logit = -INFINITY;
  if (j < NACT) {
    float acc = mb[j];
    const float* hr = hidden + b * 512;
    const float* wr = mw + j * 512;
    for (int k = 0; k < 512; ++k) acc += hr[k] * wr[k];
    logit = acc;
  }
  float m = logit;
#pragma unroll
  for (int off = 16; off >= 1; off >>= 1) m = fmaxf(m, __shfl_xor(m, off, 32));
  float e = (j < NACT) ? __expf(logit - m) : 0.f;
  float s = e;
#pragma unroll
  for (int off = 16; off >= 1; off >>= 1) s += __shfl_xor(s, off, 32);
  float logp = logit - m - __logf(s);
  int act = action[b];
  if (j == act) out[BB + NAG + b] = logp;  // log_prob slot
  float ent = (j < NACT) ? -__expf(logp) * logp : 0.f;
#pragma unroll
  for (int off = 16; off >= 1; off >>= 1) ent += __shfl_xor(ent, off, 32);
  if (j == 0) {
    out[BB + NAG + BB + b] = ent;  // entropy slot
    out[b] = (float)act;           // action passthrough
  }
}

// ---------------------------------------------------------------------------
// enc0[n] = hidden[0,:] . msg_w[n,:] + msg_b[n]   (only row 0 is ever used)
// ---------------------------------------------------------------------------
__global__ void enc0_kernel(const float* __restrict__ hidden, const float* __restrict__ mw,
                            const float* __restrict__ mb, float* __restrict__ enc0) {
  int n = blockIdx.x * blockDim.x + threadIdx.x;
  if (n >= NAG) return;
  const float4* h = (const float4*)hidden;
  const float4* w = (const float4*)(mw + n * 512);
  float acc = mb[n];
  for (int k = 0; k < 128; ++k) {
    float4 a = h[k], c = w[k];
    acc += a.x * c.x + a.y * c.y + a.z * c.z + a.w * c.w;
  }
  enc0[n] = acc;
}

// ---------------------------------------------------------------------------
// GCN normalization: deg[d] = 1 (self loop) + #edges into d; dinv = rsqrt(deg)
// ---------------------------------------------------------------------------
__global__ void deg_init_kernel(float* __restrict__ deg) {
  int n = blockIdx.x * blockDim.x + threadIdx.x;
  if (n < NAG) deg[n] = 1.0f;
}
__global__ void deg_edge_kernel(const int* __restrict__ ei, float* __restrict__ deg) {
  int e = blockIdx.x * blockDim.x + threadIdx.x;
  if (e < NE) atomicAdd(&deg[ei[2 * e + 1]], 1.0f);
}
__global__ void dinv_kernel(const float* __restrict__ deg, float* __restrict__ dinv) {
  int n = blockIdx.x * blockDim.x + threadIdx.x;
  if (n < NAG) dinv[n] = rsqrtf(deg[n]);  // deg >= 1 always
}

// ---------------------------------------------------------------------------
// GCN1: hw[n,k] = x_msg[n]*W1[0,k] + enc0[n]*W1[1,k]; ag1 seeded w/ self-loop
// ---------------------------------------------------------------------------
__global__ void gcn1_init_kernel(const float* __restrict__ x_msg, const float* __restrict__ enc0,
                                 const float* __restrict__ w1, const float* __restrict__ dinv,
                                 float* __restrict__ hw, float* __restrict__ ag1) {
  int n = blockIdx.x * blockDim.x + threadIdx.x;
  if (n >= NAG) return;
  float f0 = x_msg[n], f1 = enc0[n];
  float dn = dinv[n], self = dn * dn;
#pragma unroll
  for (int k = 0; k < 16; ++k) {
    float v = f0 * w1[k] + f1 * w1[16 + k];
    hw[n * 16 + k] = v;
    ag1[n * 16 + k] = self * v;
  }
}

// generic edge aggregation over 16-wide features: acc[d] += dinv[s]*dinv[d]*feat[s]
__global__ void edge_agg_kernel(const int* __restrict__ ei, const float* __restrict__ dinv,
                                const float* __restrict__ feat, float* __restrict__ acc) {
  int e = blockIdx.x * blockDim.x + threadIdx.x;
  if (e >= NE) return;
  int s = ei[2 * e], d = ei[2 * e + 1];
  float nrm = dinv[s] * dinv[d];
  const float* fs = feat + s * 16;
  float* ad = acc + d * 16;
#pragma unroll
  for (int k = 0; k < 16; ++k) atomicAdd(&ad[k], nrm * fs[k]);
}

__global__ void gcn1_fin_kernel(const float* __restrict__ ag1, const float* __restrict__ b1,
                                float* __restrict__ g) {
  int t = blockIdx.x * blockDim.x + threadIdx.x;
  if (t < NAG * 16) g[t] = fmaxf(ag1[t] + b1[t & 15], 0.f);
}

__global__ void ag2_init_kernel(const float* __restrict__ g, const float* __restrict__ dinv,
                                float* __restrict__ ag2) {
  int t = blockIdx.x * blockDim.x + threadIdx.x;
  if (t >= NAG * 16) return;
  float dn = dinv[t >> 4];
  ag2[t] = dn * dn * g[t];
}

// ---------------------------------------------------------------------------
// Fused GCN2 + log_softmax-mean:
//   out[i,:] = ag2[i,:16] @ W2[16,8192] + b2   (never materialized; 256 MB saved)
//   msg_out[i] = mean_j(out) - logsumexp_j(out)
// 8 waves/block, each owns a 16-row tile. W2 is staged per block into LDS in
// 16x256 chunks (double-buffered, GLOBAL_LOAD_ASYNC_TO_LDS_B128 / ASYNCcnt),
// cutting L2 traffic 8x vs per-wave streaming. WMMA f32 16x16x4 chained over
// K=16; per-lane online (max,sumexp,sum); shfl_xor reduce across 16 lanes.
// RSTRIDE=264 floats: rows land on distinct LDS banks for the B-frag pattern.
// ---------------------------------------------------------------------------
#define RSTRIDE 264
#define CHUNK   256
#define NCHUNK  (NAG / CHUNK)

__global__ void __launch_bounds__(256)
gcn2_lse_kernel(const float* __restrict__ ag2, const float* __restrict__ w2,
                const float* __restrict__ b2, float* __restrict__ msg_out) {
  __shared__ __align__(16) float sbuf[2][16 * RSTRIDE];
  __shared__ float sbias[2][CHUNK];

  int wid  = threadIdx.x >> 5;
  int lane = threadIdx.x & 31;
  int half = lane >> 4, lid = lane & 15;
  int m0 = (blockIdx.x * 8 + wid) * 16;

#if HAVE_WMMA_F32X4
  // A (16x4 f32) layout: lanes 0-15 hold K=4c+{0,1}; lanes 16-31 hold K=4c+{2,3}
  v2f afrag[4];
#pragma unroll
  for (int c = 0; c < 4; ++c) {
    int arow = m0 + lid;
    afrag[c][0] = ag2[arow * 16 + 4 * c + 2 * half];
    afrag[c][1] = ag2[arow * 16 + 4 * c + 2 * half + 1];
  }
#endif

  float m[8], s[8], t[8];
#pragma unroll
  for (int r = 0; r < 8; ++r) { m[r] = -INFINITY; s[r] = 0.f; t[r] = 0.f; }

#if HAVE_ASYNC_LDS
  // ---- async double-buffered staging: 4x b128 + 1x b32 per thread per chunk
  {
    int tt = threadIdx.x;
#pragma unroll
    for (int q = 0; q < 4; ++q) {
      int flat = q * 256 + tt;
      int k = flat >> 6, off = (flat & 63) * 4;
      __builtin_amdgcn_global_load_async_to_lds_b128(
          as_g4(&w2[k * NAG + off]), as_l4(&sbuf[0][k * RSTRIDE + off]), 0, 0);
    }
    __builtin_amdgcn_global_load_async_to_lds_b32(
        as_g1(&b2[tt]), as_l1(&sbias[0][tt]), 0, 0);
  }
#endif

  for (int ci = 0; ci < NCHUNK; ++ci) {
    int c0 = ci * CHUNK;
#if HAVE_ASYNC_LDS
    if (ci + 1 < NCHUNK) {  // issue next chunk while current one is in flight
      int tt = threadIdx.x;
      int c1 = c0 + CHUNK, bsel = (ci + 1) & 1;
#pragma unroll
      for (int q = 0; q < 4; ++q) {
        int flat = q * 256 + tt;
        int k = flat >> 6, off = (flat & 63) * 4;
        __builtin_amdgcn_global_load_async_to_lds_b128(
            as_g4(&w2[k * NAG + c1 + off]), as_l4(&sbuf[bsel][k * RSTRIDE + off]), 0, 0);
      }
      __builtin_amdgcn_global_load_async_to_lds_b32(
          as_g1(&b2[c1 + tt]), as_l1(&sbias[bsel][tt]), 0, 0);
      WAIT_ASYNC(5);  // own wave's chunk-ci loads landed; ci+1 still in flight
    } else {
      WAIT_ASYNC(0);
    }
    __syncthreads();  // all waves' chunk-ci data visible
    const float* buf = sbuf[ci & 1];
    const float* bb  = sbias[ci & 1];
#else
    // ---- fallback: cooperative synchronous staging (single buffer)
    float4 tmpw[4];
    float  tmpb;
    {
      int tt = threadIdx.x;
#pragma unroll
      for (int q = 0; q < 4; ++q) {
        int flat = q * 256 + tt;
        int k = flat >> 6, off = (flat & 63) * 4;
        tmpw[q] = *(const float4*)&w2[k * NAG + c0 + off];
      }
      tmpb = b2[c0 + tt];
    }
    __syncthreads();  // previous chunk fully consumed before overwrite
    {
      int tt = threadIdx.x;
#pragma unroll
      for (int q = 0; q < 4; ++q) {
        int flat = q * 256 + tt;
        int k = flat >> 6, off = (flat & 63) * 4;
        *(float4*)&sbuf[0][k * RSTRIDE + off] = tmpw[q];
      }
      sbias[0][tt] = tmpb;
    }
    __syncthreads();
    const float* buf = sbuf[0];
    const float* bb  = sbias[0];
#endif

#pragma unroll 4
    for (int u = 0; u < 16; ++u) {
      v8f d = {};
#if HAVE_WMMA_F32X4
#pragma unroll
      for (int c = 0; c < 4; ++c) {
        v2f bfrag;  // B (4x16 f32): lanes 0-15 K=4c+{0,1}; lanes 16-31 K=4c+{2,3}
        bfrag[0] = buf[(4 * c + 2 * half) * RSTRIDE + u * 16 + lid];
        bfrag[1] = buf[(4 * c + 2 * half + 1) * RSTRIDE + u * 16 + lid];
        d = __builtin_amdgcn_wmma_f32_16x16x4_f32(false, afrag[c], false, bfrag,
                                                  (short)0, d, false, false);
      }
#else
#pragma unroll
      for (int r = 0; r < 8; ++r) {
        int row = m0 + r + half * 8;
        float acc = 0.f;
#pragma unroll
        for (int k = 0; k < 16; ++k)
          acc += ag2[row * 16 + k] * buf[k * RSTRIDE + u * 16 + lid];
        d[r] = acc;
      }
#endif
      float bias = bb[u * 16 + lid];
#pragma unroll
      for (int r = 0; r < 8; ++r) {
        float x = d[r] + bias;
        t[r] += x;
        if (x > m[r]) { s[r] = s[r] * __expf(m[r] - x) + 1.f; m[r] = x; }
        else          { s[r] += __expf(x - m[r]); }
      }
    }
    __syncthreads();  // chunk consumed; safe to overwrite next iteration
  }

  // reduce (m,s,t) across the 16 lanes of each half (same rows, disjoint cols)
#pragma unroll
  for (int off = 1; off < 16; off <<= 1) {
#pragma unroll
    for (int r = 0; r < 8; ++r) {
      float om = __shfl_xor(m[r], off, 32);
      float os = __shfl_xor(s[r], off, 32);
      float ot = __shfl_xor(t[r], off, 32);
      float nm = fmaxf(m[r], om);
      s[r] = s[r] * __expf(m[r] - nm) + os * __expf(om - nm);
      m[r] = nm;
      t[r] += ot;
    }
  }
  if (lid == 0) {
#pragma unroll
    for (int r = 0; r < 8; ++r) {
      int row = m0 + r + half * 8;
      msg_out[row] = t[r] * (1.0f / (float)NAG) - (m[r] + __logf(s[r]));
    }
  }
}

// ---------------------------------------------------------------------------
// launch
// ---------------------------------------------------------------------------
extern "C" void kernel_launch(void* const* d_in, const int* in_sizes, int n_in,
                              void* d_out, int out_size, void* d_ws, size_t ws_size,
                              hipStream_t stream) {
  const float* x       = (const float*)d_in[0];
  const float* x_msg   = (const float*)d_in[1];
  const int*   eidx    = (const int*)  d_in[2];
  const int*   action  = (const int*)  d_in[3];
  const float* conv1_w = (const float*)d_in[4];
  const float* conv1_b = (const float*)d_in[5];
  const float* conv2_w = (const float*)d_in[6];
  const float* conv2_b = (const float*)d_in[7];
  const float* fc_w    = (const float*)d_in[8];
  const float* fc_b    = (const float*)d_in[9];
  const float* fcmu_w  = (const float*)d_in[10];
  const float* fcmu_b  = (const float*)d_in[11];
  const float* msg_w   = (const float*)d_in[12];
  const float* msg_b   = (const float*)d_in[13];
  const float* gcn1_w  = (const float*)d_in[14];
  const float* gcn1_b  = (const float*)d_in[15];
  const float* gcn2_w  = (const float*)d_in[16];
  const float* gcn2_b  = (const float*)d_in[17];
  float* out = (float*)d_out;

  // workspace layout (floats)
  float* ws = (float*)d_ws;
  float* h1     = ws;                 // 294912
  float* h2     = h1  + 294912;       // 262144
  float* hidden = h2  + 262144;       // 131072
  float* enc0   = hidden + 131072;    // 8192
  float* deg    = enc0 + NAG;         // 8192
  float* dinv   = deg  + NAG;         // 8192
  float* hw1    = dinv + NAG;         // 131072
  float* ag1    = hw1  + NAG * 16;    // 131072
  float* g      = ag1  + NAG * 16;    // 131072
  float* ag2    = g    + NAG * 16;    // 131072  (total ~4.95 MB)

  conv1_kernel<<<(BB * 32 * 36 + 255) / 256, 256, 0, stream>>>(x, conv1_w, conv1_b, h1);
  conv2_kernel<<<(BB * 64 * 16 + 255) / 256, 256, 0, stream>>>(h1, conv2_w, conv2_b, h2);
  fc_wmma_kernel<<<64, 256, 0, stream>>>(h2, fc_w, fc_b, hidden);  // 512 waves, 1 tile each
  head_kernel<<<BB, 32, 0, stream>>>(hidden, fcmu_w, fcmu_b, action, out);
  enc0_kernel<<<(NAG + 255) / 256, 256, 0, stream>>>(hidden, msg_w, msg_b, enc0);

  deg_init_kernel<<<(NAG + 255) / 256, 256, 0, stream>>>(deg);
  deg_edge_kernel<<<(NE + 255) / 256, 256, 0, stream>>>(eidx, deg);
  dinv_kernel<<<(NAG + 255) / 256, 256, 0, stream>>>(deg, dinv);

  gcn1_init_kernel<<<(NAG + 255) / 256, 256, 0, stream>>>(x_msg, enc0, gcn1_w, dinv, hw1, ag1);
  edge_agg_kernel<<<(NE + 255) / 256, 256, 0, stream>>>(eidx, dinv, hw1, ag1);
  gcn1_fin_kernel<<<(NAG * 16 + 255) / 256, 256, 0, stream>>>(ag1, gcn1_b, g);

  ag2_init_kernel<<<(NAG * 16 + 255) / 256, 256, 0, stream>>>(g, dinv, ag2);
  edge_agg_kernel<<<(NE + 255) / 256, 256, 0, stream>>>(eidx, dinv, g, ag2);

  gcn2_lse_kernel<<<64, 256, 0, stream>>>(ag2, gcn2_w, gcn2_b, out + BB);
}
